// BasicBlock_24850680775269
// MI455X (gfx1250) — 8-bit matrix operand order repaired and checked by reading, compiled, not run on hardware
//
#include <hip/hip_runtime.h>

typedef int v8i __attribute__((ext_vector_type(8)));
typedef int v4i __attribute__((ext_vector_type(4)));
typedef int v2i __attribute__((ext_vector_type(2)));

#define BATCH 32
#define CH    256
#define HH    56
#define WW    56
#define HP    58
#define WP    58
#define KTOT  2304
#define NPIX  (HH * WW)
#define TOTAL (BATCH * CH * NPIX)

__global__ void pack_weights(const float* __restrict__ w, signed char* __restrict__ wq) {
  int o = blockIdx.x * blockDim.x + threadIdx.x;
  if (o >= CH * KTOT) return;
  int ci   = o & 255;
  int khkw = (o >> 8) % 9;
  int co   = o / KTOT;
  int kh = khkw / 3, kw = khkw - kh * 3;
  float v = w[((co * CH + ci) * 3 + kh) * 3 + kw];
  wq[o] = (v >= 0.0f) ? (signed char)1 : (signed char)-1;
}

__global__ void pack_acts(const float* __restrict__ x, signed char* __restrict__ xq) {
  int idx = blockIdx.x * blockDim.x + threadIdx.x;
  if (idx >= BATCH * HP * WP * CH) return;
  int ci = idx & 255;
  int r  = idx >> 8;
  int wp = r % WP; r /= WP;
  int hp = r % HP;
  int b  = r / HP;
  signed char v = -1;
  if (hp >= 1 && hp <= HH && wp >= 1 && wp <= WW) {
    float xv = x[((b * CH + ci) * HH + (hp - 1)) * WW + (wp - 1)];
    v = (xv >= 0.0f) ? (signed char)1 : (signed char)-1;
  }
  xq[idx] = v;
}

__global__ __launch_bounds__(256) void bconv_wmma(
    const signed char* __restrict__ wq,
    const signed char* __restrict__ xq,
    float* __restrict__ out) {
  __shared__ __align__(16) signed char Bsh[3 * WP * CH];

  const int lane = threadIdx.x & 31;
  const int wave = threadIdx.x >> 5;
  const int half = lane >> 4;
  const int nlo  = lane & 15;

  const int oh = blockIdx.x % HH;
  const int b  = blockIdx.x / HH;

  const int m_base = wave * 32;

  {
    const int* __restrict__ src = (const int*)(xq + (size_t)((b * HP + oh) * WP) * CH);
    int* dst = (int*)Bsh;
    for (int i = threadIdx.x; i < (3 * WP * CH) / 4; i += 256) dst[i] = src[i];
  }
  __syncthreads();

  v8i acc[2][4];
#pragma unroll
  for (int mt = 0; mt < 2; ++mt)
#pragma unroll
    for (int nt = 0; nt < 4; ++nt)
      acc[mt][nt] = (v8i){};

  const int m0 = m_base + nlo;
  const int m1 = m_base + 16 + nlo;

  const v2i* __restrict__ wq64  = (const v2i*)wq;
  const v4i* __restrict__ bsh128 = (const v4i*)Bsh;

  for (int kc = 0; kc < 36; ++kc) {
    const int khkw  = kc >> 2;
    const int c0    = (kc & 3) << 6;
    const int kh    = khkw / 3;
    const int kw    = khkw - kh * 3;
    const int kbase = kc << 6;

    v8i a0, a1;
#pragma unroll
    for (int g = 0; g < 4; ++g) {
      int kOff = (g >> 1) * 32 + half * 16 + (g & 1) * 8;
      v2i t0 = wq64[(m0 * KTOT + kbase + kOff) >> 3];
      v2i t1 = wq64[(m1 * KTOT + kbase + kOff) >> 3];
      a0[2 * g] = t0.x; a0[2 * g + 1] = t0.y;
      a1[2 * g] = t1.x; a1[2 * g + 1] = t1.y;
    }

    v8i bv[4];
#pragma unroll
    for (int nt = 0; nt < 4; ++nt) {
      int col = nt * 16 + nlo + kw;
      if (col > WP - 1) col = WP - 1;
      int off = (kh * WP + col) * CH + c0 + half * 16;
      v4i lo = bsh128[off >> 4];
      v4i hi = bsh128[(off + 32) >> 4];
      bv[nt][0] = lo.x; bv[nt][1] = lo.y; bv[nt][2] = lo.z; bv[nt][3] = lo.w;
      bv[nt][4] = hi.x; bv[nt][5] = hi.y; bv[nt][6] = hi.z; bv[nt][7] = hi.w;
    }

#pragma unroll
    for (int nt = 0; nt < 4; ++nt) {
      acc[0][nt] = __builtin_amdgcn_wmma_i32_16x16x64_iu8(true, a0, true, bv[nt],
                                                          acc[0][nt], false, false);
      acc[1][nt] = __builtin_amdgcn_wmma_i32_16x16x64_iu8(true, a1, true, bv[nt],
                                                          acc[1][nt], false, false);
    }
  }

#pragma unroll
  for (int nt = 0; nt < 4; ++nt) {
    int ow = nt * 16 + nlo;
    if (ow < WW) {
      int outcol = oh * WW + ow;
#pragma unroll
      for (int v = 0; v < 8; ++v) {
        int co0 = m_base + v + 8 * half;
        out[(b * CH + co0) * NPIX + outcol]        = (float)acc[0][nt][v];
        out[(b * CH + (co0 + 16)) * NPIX + outcol] = (float)acc[1][nt][v];
      }
    }
  }
}

__global__ __launch_bounds__(256) void chan_stats(const float* __restrict__ conv,
                                                  long long* __restrict__ sums,
                                                  long long* __restrict__ sumsq) {
  __shared__ long long s1[256];
  __shared__ long long s2[256];
  int c = blockIdx.x;
  int t = threadIdx.x;
  long long a1 = 0, a2 = 0;
  for (int i = t; i < BATCH * NPIX; i += 256) {
    int bb = i / NPIX;
    int s  = i - bb * NPIX;
    int iv = (int)conv[(bb * CH + c) * NPIX + s];
    a1 += iv;
    a2 += (long long)iv * (long long)iv;
  }
  s1[t] = a1; s2[t] = a2;
  __syncthreads();
  for (int off = 128; off > 0; off >>= 1) {
    if (t < off) { s1[t] += s1[t + off]; s2[t] += s2[t + off]; }
    __syncthreads();
  }
  if (t == 0) { sums[c] = s1[0]; sumsq[c] = s2[0]; }
}

__global__ void bn_coeffs(const long long* __restrict__ sums,
                          const long long* __restrict__ sumsq,
                          const float* __restrict__ gamma,
                          const float* __restrict__ beta,
                          float* __restrict__ scale,
                          float* __restrict__ shift) {
  int c = threadIdx.x;
  const double cnt = (double)(BATCH * NPIX);
  double m   = (double)sums[c] / cnt;
  double var = (double)sumsq[c] / cnt - m * m;
  float sc = gamma[c] * rsqrtf((float)var + 1e-5f);
  scale[c] = sc;
  shift[c] = beta[c] - (float)m * sc;
}

__global__ void bn_residual(const float* __restrict__ x,
                            const float* __restrict__ scale,
                            const float* __restrict__ shift,
                            float* __restrict__ out) {
  int i = blockIdx.x * blockDim.x + threadIdx.x;
  if (i >= TOTAL) return;
  int c = (i / NPIX) & 255;
  out[i] = out[i] * scale[c] + shift[c] + x[i];
}

extern "C" void kernel_launch(void* const* d_in, const int* in_sizes, int n_in,
                              void* d_out, int out_size, void* d_ws, size_t ws_size,
                              hipStream_t stream) {
  const float* x     = (const float*)d_in[0];
  const float* w     = (const float*)d_in[1];
  const float* gamma = (const float*)d_in[2];
  const float* beta  = (const float*)d_in[3];
  float* out = (float*)d_out;

  char* ws = (char*)d_ws;
  const size_t WQ_BYTES = (size_t)CH * KTOT;
  const size_t XQ_BYTES = (size_t)BATCH * HP * WP * CH;
  signed char* wq   = (signed char*)ws;
  signed char* xq   = (signed char*)(ws + WQ_BYTES);
  long long*   sums  = (long long*)(ws + WQ_BYTES + XQ_BYTES);
  long long*   sumsq = sums + 256;
  float*       scale = (float*)(ws + WQ_BYTES + XQ_BYTES + 4096);
  float*       shift = scale + 256;

  pack_weights<<<(CH * KTOT + 255) / 256, 256, 0, stream>>>(w, wq);
  pack_acts<<<(BATCH * HP * WP * CH + 255) / 256, 256, 0, stream>>>(x, xq);
  bconv_wmma<<<BATCH * HH, 256, 0, stream>>>(wq, xq, out);
  chan_stats<<<CH, 256, 0, stream>>>(out, sums, sumsq);
  bn_coeffs<<<1, 256, 0, stream>>>(sums, sumsq, gamma, beta, scale, shift);
  bn_residual<<<(TOTAL + 255) / 256, 256, 0, stream>>>(x, scale, shift, out);
}
